// LongformerSelfAttention_35416300323374
// MI455X (gfx1250) — matrix operand order repaired and checked by reading, not compiled, not run on hardware
//
#include <hip/hip_runtime.h>

// ---------------------------------------------------------------------------
// Longformer-style self attention for MI455X (gfx1250, wave32, WMMA).
//
// Roofline: ~60 GFLOP vs ~46 MB of traffic (2us @ 23.3 TB/s) -> compute bound,
// so all matmuls go through v_wmma_f32_16x16x32_f16 (f16 in, f32 accum).
// global_bias is a per-head scalar added to every score of a softmax row ->
// cancels exactly (softmax shift invariance) and is dropped. The combined
// scale log2(e)/sqrt(D) is folded into the stored f16 Q so the streaming
// softmax runs in the base-2 domain (v_exp_f32 is natively base-2).
// ---------------------------------------------------------------------------

typedef __attribute__((ext_vector_type(16))) _Float16 v16h;
typedef __attribute__((ext_vector_type(8)))  _Float16 v8h;
typedef __attribute__((ext_vector_type(4)))  _Float16 v4h;
typedef __attribute__((ext_vector_type(8)))  float    v8f;

#define NB   2
#define NS   2048
#define ND   1024
#define NH   16
#define NHD  64
#define NBH  (NB * NH)            // 32

// log2(e) / sqrt(1024)
#define QSCALE 0.0450842187778112f

static __device__ __forceinline__ v8f wmma_f16(v16h a, v16h b, v8f c) {
    return __builtin_amdgcn_wmma_f32_16x16x32_f16(
        /*neg_a=*/false, a, /*neg_b=*/false, b,
        /*c_mod=*/(short)0, c, /*reuse_a=*/false, /*reuse_b=*/false);
}

// ---------------------------------------------------------------------------
// Kernel 1: convert x and Wq/Wk/Wv (fp32) to f16 in workspace.
// xh: [B*S][D] row-major, wh: [3][D_out][D_in] row-major (K-contiguous rows).
// ---------------------------------------------------------------------------
__global__ void __launch_bounds__(256) k_convert(
    const float* __restrict__ x,
    const float* __restrict__ Wq,
    const float* __restrict__ Wk,
    const float* __restrict__ Wv,
    _Float16* __restrict__ xh,
    _Float16* __restrict__ wh)
{
    const int tid  = blockIdx.x * blockDim.x + threadIdx.x;
    const int nthr = gridDim.x * blockDim.x;
    const int XV = (NB * NS * ND) / 4;  // 1048576
    const int WV = (ND * ND) / 4;       // 262144

    for (int i = tid; i < XV; i += nthr) {
        float4 f = ((const float4*)x)[i];
        v4h o = {(_Float16)f.x, (_Float16)f.y, (_Float16)f.z, (_Float16)f.w};
        ((v4h*)xh)[i] = o;
    }
    for (int i = tid; i < WV; i += nthr) {
        float4 f = ((const float4*)Wq)[i];
        v4h o = {(_Float16)f.x, (_Float16)f.y, (_Float16)f.z, (_Float16)f.w};
        ((v4h*)wh)[i] = o;
    }
    for (int i = tid; i < WV; i += nthr) {
        float4 f = ((const float4*)Wk)[i];
        v4h o = {(_Float16)f.x, (_Float16)f.y, (_Float16)f.z, (_Float16)f.w};
        ((v4h*)(wh + (size_t)ND * ND))[i] = o;
    }
    for (int i = tid; i < WV; i += nthr) {
        float4 f = ((const float4*)Wv)[i];
        v4h o = {(_Float16)f.x, (_Float16)f.y, (_Float16)f.z, (_Float16)f.w};
        ((v4h*)(wh + (size_t)2 * ND * ND))[i] = o;
    }
}

// ---------------------------------------------------------------------------
// Kernel 2: QKV projection GEMM: y = xh @ W^T + b  (nn.Linear semantics).
// Per-wave tile: 32(M) x 64(N), K-loop step 32, 8 WMMAs per step.
// Outputs:
//   mat 0 -> qh  [BH][S][hd]  f16, pre-scaled by log2(e)/32
//   mat 1 -> kh  [BH][S][hd]  f16
//   mat 2 -> vth [BH][hd][S]  f16 (transposed; C-layout makes this the
//                                  contiguous b128 store path)
// ---------------------------------------------------------------------------
__global__ void __launch_bounds__(256) k_qkv_gemm(
    const _Float16* __restrict__ xh,
    const _Float16* __restrict__ wh,
    const float* __restrict__ bq,
    const float* __restrict__ bk,
    const float* __restrict__ bv,
    _Float16* __restrict__ qh,
    _Float16* __restrict__ kh,
    _Float16* __restrict__ vth)
{
    const int lane = threadIdx.x & 31;
    const int hs   = lane >> 4;     // half-wave select
    const int ln   = lane & 15;
    const int wave = (blockIdx.x * blockDim.x + threadIdx.x) >> 5;  // 0..6143

    const int mat    = wave >> 11;          // 0..2  (2048 tiles per matrix)
    const int rem    = wave & 2047;
    const int tile_m = rem >> 4;            // 0..127
    const int tile_n = rem & 15;            // 0..15
    const int m0 = tile_m * 32;
    const int n0 = tile_n * 64;

    const _Float16* w    = wh + (size_t)mat * ND * ND;
    const float*    bias = (mat == 0) ? bq : ((mat == 1) ? bk : bv);

    v8f c[2][4];
#pragma unroll
    for (int tm = 0; tm < 2; ++tm)
#pragma unroll
        for (int tn = 0; tn < 4; ++tn) c[tm][tn] = (v8f)0.0f;

    const _Float16* arow[2];
#pragma unroll
    for (int tm = 0; tm < 2; ++tm)
        arow[tm] = xh + (size_t)(m0 + tm * 16 + ln) * ND;
    const _Float16* brow[4];
#pragma unroll
    for (int tn = 0; tn < 4; ++tn)
        brow[tn] = w + (size_t)(n0 + tn * 16 + ln) * ND;

    for (int k0 = 0; k0 < ND; k0 += 32) {
        v16h a[2];
#pragma unroll
        for (int tm = 0; tm < 2; ++tm) {
            v8h lo = *(const v8h*)(arow[tm] + k0 + hs * 16);
            v8h hi = *(const v8h*)(arow[tm] + k0 + hs * 16 + 8);
#pragma unroll
            for (int i = 0; i < 8; ++i) { a[tm][i] = lo[i]; a[tm][i + 8] = hi[i]; }
        }
        v16h b[4];
#pragma unroll
        for (int tn = 0; tn < 4; ++tn)
            b[tn] = *(const v16h*)(brow[tn] + k0 + hs * 16);
#pragma unroll
        for (int tm = 0; tm < 2; ++tm)
#pragma unroll
            for (int tn = 0; tn < 4; ++tn)
                c[tm][tn] = wmma_f16(a[tm], b[tn], c[tm][tn]);
    }

    // Epilogue: bias add, (q) scale, store in attention-friendly layouts.
#pragma unroll
    for (int tn = 0; tn < 4; ++tn) {
        const int n = n0 + tn * 16 + ln;   // output feature (per lane)
        const float bval = bias[n];
        const int h = n >> 6;
        const int e = n & 63;
#pragma unroll
        for (int tm = 0; tm < 2; ++tm) {
            const int mrow0 = m0 + tm * 16 + 8 * hs;   // first of 8 rows
            const int bidx  = mrow0 >> 11;             // batch
            const int srow0 = mrow0 & 2047;            // sequence pos
            const int bh    = bidx * NH + h;
            if (mat == 2) {
                v8h o;
#pragma unroll
                for (int r = 0; r < 8; ++r)
                    o[r] = (_Float16)(c[tm][tn][r] + bval);
                *(v8h*)(vth + ((size_t)bh * NHD + e) * NS + srow0) = o;
            } else {
                _Float16* dst = (mat == 0) ? qh : kh;
                const float scl = (mat == 0) ? QSCALE : 1.0f;
#pragma unroll
                for (int r = 0; r < 8; ++r)
                    dst[((size_t)bh * NS + srow0 + r) * NHD + e] =
                        (_Float16)((c[tm][tn][r] + bval) * scl);
            }
        }
    }
}

// ---------------------------------------------------------------------------
// Kernel 3: flash attention. 1 wave = 16 queries of one (b,h); streams 64-key
// blocks: 8 score WMMAs + base-2 online softmax (16-lane shfl reductions) +
// P C->A layout shuffle through per-wave LDS + 8 P@V WMMAs vs transposed V.
// ---------------------------------------------------------------------------
__global__ void __launch_bounds__(128) k_attn(
    const _Float16* __restrict__ qh,
    const _Float16* __restrict__ kh,
    const _Float16* __restrict__ vth,
    float* __restrict__ out)
{
    __shared__ __align__(16) _Float16 lds_p[4][16][72];  // per-wave P buf (padded)

    const int lane = threadIdx.x & 31;
    const int hs   = lane >> 4;
    const int ln   = lane & 15;
    const int w    = threadIdx.x >> 5;      // wave in block (0..3)
    const int bh   = blockIdx.x >> 5;       // 0..31
    const int qb   = blockIdx.x & 31;       // 0..31
    const int q0   = (qb * 4 + w) * 16;     // query tile start

    const _Float16* qbase = qh  + (size_t)bh * NS * NHD;
    const _Float16* kbase = kh  + (size_t)bh * NS * NHD;
    const _Float16* vbase = vth + (size_t)bh * NHD * NS;

    // Q A-tiles: hd slices [0,32) and [32,64); pre-scaled by log2(e)/32.
    v16h aq[2];
    {
        const _Float16* qrow = qbase + (size_t)(q0 + ln) * NHD;
#pragma unroll
        for (int t = 0; t < 2; ++t) {
            v8h lo = *(const v8h*)(qrow + t * 32 + hs * 16);
            v8h hi = *(const v8h*)(qrow + t * 32 + hs * 16 + 8);
#pragma unroll
            for (int i = 0; i < 8; ++i) { aq[t][i] = lo[i]; aq[t][i + 8] = hi[i]; }
        }
    }

    v8f o[4];
#pragma unroll
    for (int t = 0; t < 4; ++t) o[t] = (v8f)0.0f;
    float mi[8], li[8];
#pragma unroll
    for (int r = 0; r < 8; ++r) { mi[r] = -INFINITY; li[r] = 0.0f; }

    const bool mtl = (q0 == 0);        // top-left corner mask tile owner
    const bool mbr = (q0 == NS - 16);  // bottom-right corner mask tile owner

    for (int kb = 0; kb < NS; kb += 64) {
        // Prefetch next key block's K and V rows into cache.
        if (kb + 64 < NS) {
            __builtin_prefetch(kbase + (size_t)(kb + 64 + lane) * NHD, 0, 3);
            __builtin_prefetch(vbase + (size_t)(lane << 1) * NS + kb + 64, 0, 3);
        }

        // ---- scores: four 16x16 tiles (keys kb+16*kn .. +15) ----
        v8f c[4];
#pragma unroll
        for (int kn = 0; kn < 4; ++kn) c[kn] = (v8f)0.0f;
#pragma unroll
        for (int kn = 0; kn < 4; ++kn) {
            const _Float16* kr = kbase + (size_t)(kb + kn * 16 + ln) * NHD;
            v16h b0 = *(const v16h*)(kr + hs * 16);
            v16h b1 = *(const v16h*)(kr + 32 + hs * 16);
            c[kn] = wmma_f16(aq[0], b0, c[kn]);
            c[kn] = wmma_f16(aq[1], b1, c[kn]);
        }

        // ---- corner mask: rows<8 & keys<8, rows>=2040 & keys>=2040 ----
        if (mtl && kb == 0 && hs == 0 && ln < 8) {
#pragma unroll
            for (int r = 0; r < 8; ++r) c[0][r] = -INFINITY;
        }
        if (mbr && kb == NS - 64 && hs == 1 && ln >= 8) {
#pragma unroll
            for (int r = 0; r < 8; ++r) c[3][r] = -INFINITY;
        }

        // ---- base-2 online softmax over these 64 columns ----
#pragma unroll
        for (int r = 0; r < 8; ++r) {
            float mx = fmaxf(fmaxf(c[0][r], c[1][r]), fmaxf(c[2][r], c[3][r]));
            mx = fmaxf(mx, __shfl_xor(mx, 1, 32));
            mx = fmaxf(mx, __shfl_xor(mx, 2, 32));
            mx = fmaxf(mx, __shfl_xor(mx, 4, 32));
            mx = fmaxf(mx, __shfl_xor(mx, 8, 32));
            const float mn = fmaxf(mi[r], mx);
            const float sc = exp2f(mi[r] - mn);
            mi[r] = mn;
            const float e0 = exp2f(c[0][r] - mn);
            const float e1 = exp2f(c[1][r] - mn);
            const float e2 = exp2f(c[2][r] - mn);
            const float e3 = exp2f(c[3][r] - mn);
            float rs = (e0 + e1) + (e2 + e3);
            rs += __shfl_xor(rs, 1, 32);
            rs += __shfl_xor(rs, 2, 32);
            rs += __shfl_xor(rs, 4, 32);
            rs += __shfl_xor(rs, 8, 32);
            li[r] = li[r] * sc + rs;
#pragma unroll
            for (int t = 0; t < 4; ++t) o[t][r] *= sc;
            // stage P (C-layout element -> [row][col] in LDS)
            const int row = r + 8 * hs;
            lds_p[w][row][ln]      = (_Float16)e0;
            lds_p[w][row][16 + ln] = (_Float16)e1;
            lds_p[w][row][32 + ln] = (_Float16)e2;
            lds_p[w][row][48 + ln] = (_Float16)e3;
        }
        asm volatile("s_wait_dscnt 0" ::: "memory");

        // ---- re-read P in WMMA A-layout: two 16x32 tiles ----
        v16h ap[2];
#pragma unroll
        for (int a = 0; a < 2; ++a) {
            const _Float16* prow = &lds_p[w][ln][a * 32];
            v8h lo = *(const v8h*)(prow + hs * 16);
            v8h hi = *(const v8h*)(prow + hs * 16 + 8);
#pragma unroll
            for (int i = 0; i < 8; ++i) { ap[a][i] = lo[i]; ap[a][i + 8] = hi[i]; }
        }

        // ---- ctx accumulate: P (16x64) @ V (64x64), V stored transposed ----
#pragma unroll
        for (int t = 0; t < 4; ++t) {
#pragma unroll
            for (int a = 0; a < 2; ++a) {
                const _Float16* vrow =
                    vbase + (size_t)(t * 16 + ln) * NS + kb + a * 32 + hs * 16;
                v16h bv = *(const v16h*)vrow;
                o[t] = wmma_f16(ap[a], bv, o[t]);
            }
        }
    }

    // ---- normalize and store: out[b][s][h*64+e] fp32 ----
    const int b = bh >> 4;
    const int h = bh & 15;
#pragma unroll
    for (int t = 0; t < 4; ++t) {
        const int e = t * 16 + ln;
#pragma unroll
        for (int r = 0; r < 8; ++r) {
            const int s = q0 + r + 8 * hs;
            out[((size_t)(b * NS + s)) * ND + h * NHD + e] = o[t][r] / li[r];
        }
    }
}

// ---------------------------------------------------------------------------
// Launch. Workspace layout (bytes):
//   [0,            8388608)   xh   f16 [4096][1024]
//   [8388608,     14680064)   wh   f16 [3][1024][1024]
//   [14680064,    23068672)   qh   f16 [32][2048][64]   (pre-scaled)
//   [23068672,    31457280)   kh   f16 [32][2048][64]
//   [31457280,    39845888)   vth  f16 [32][64][2048]   (transposed V)
// ---------------------------------------------------------------------------
extern "C" void kernel_launch(void* const* d_in, const int* in_sizes, int n_in,
                              void* d_out, int out_size, void* d_ws, size_t ws_size,
                              hipStream_t stream) {
    const float* x  = (const float*)d_in[0];
    const float* Wq = (const float*)d_in[1];
    const float* bq = (const float*)d_in[2];
    const float* Wk = (const float*)d_in[3];
    const float* bk = (const float*)d_in[4];
    const float* Wv = (const float*)d_in[5];
    const float* bv = (const float*)d_in[6];
    // d_in[7] = global_bias: per-head scalar added to every score in a row;
    // cancels exactly under softmax -> intentionally unused.
    float* out = (float*)d_out;

    char* ws = (char*)d_ws;
    _Float16* xh  = (_Float16*)(ws);
    _Float16* wh  = (_Float16*)(ws + 8388608);
    _Float16* qh  = (_Float16*)(ws + 14680064);
    _Float16* kh  = (_Float16*)(ws + 23068672);
    _Float16* vth = (_Float16*)(ws + 31457280);

    k_convert<<<2048, 256, 0, stream>>>(x, Wq, Wk, Wv, xh, wh);
    // 3 matrices x (4096/32) x (1024/64) = 6144 wave-tiles, 8 waves/block
    k_qkv_gemm<<<768, 256, 0, stream>>>(xh, wh, bq, bk, bv, qh, kh, vth);
    // 32 (b,h) x 32 query-blocks; 4 waves/block, 16 queries/wave
    k_attn<<<1024, 128, 0, stream>>>(qh, kh, vth, out);
}
